// DotProductAttention_33990371180566
// MI455X (gfx1250) — matrix operand order repaired and checked by reading, not compiled, not run on hardware
//
#include <hip/hip_runtime.h>
#include <stdint.h>

typedef _Float16 v16h __attribute__((ext_vector_type(16)));
typedef _Float16 v8h  __attribute__((ext_vector_type(8)));
typedef _Float16 v4h  __attribute__((ext_vector_type(4)));
typedef float    v8f  __attribute__((ext_vector_type(8)));

#define B_      32
#define N_      2048
#define D_      128
#define QTILE   128          // query rows per block (8 waves x 16)
#define KTILE   32           // keys per iteration
#define KSTRIDE 136          // K LDS row stride in halves (272 B -> bank shift 4)
#define VSTRIDE 40           // Vt LDS row stride in halves (80 B -> bank shift 20)
#define PSTRIDE 40           // P LDS row stride in halves
#define HEAD_ELTS (1u << 18) // 2048*128 elements per head

__device__ __forceinline__ v16h cat8(v8h lo, v8h hi) {
    return __builtin_shufflevector(lo, hi, 0,1,2,3,4,5,6,7,8,9,10,11,12,13,14,15);
}

// ---------------- pre-pass: fp32 K,V -> f16 Kh (row major), Vt (per-head transposed)
__global__ __launch_bounds__(256) void cvt_kv(const float* __restrict__ K,
                                              const float* __restrict__ V,
                                              _Float16* __restrict__ Kh,
                                              _Float16* __restrict__ Vt)
{
    const size_t g   = (size_t)blockIdx.x * 256 + threadIdx.x; // 0 .. 2^21-1
    const size_t fid = g * 4;                                   // element index
    float4 kf = *(const float4*)(K + fid);
    *(v4h*)(Kh + fid) = (v4h){ (_Float16)kf.x, (_Float16)kf.y,
                               (_Float16)kf.z, (_Float16)kf.w };
    const size_t b   = fid >> 18;
    const size_t rem = fid & (HEAD_ELTS - 1);
    const int    n   = (int)(rem >> 7);
    const int    d   = (int)(rem & 127);
    float4 vf = *(const float4*)(V + fid);
    _Float16* vtb = Vt + (b << 18);                 // [d][n] within head
    vtb[(size_t)(d + 0) * N_ + n] = (_Float16)vf.x;
    vtb[(size_t)(d + 1) * N_ + n] = (_Float16)vf.y;
    vtb[(size_t)(d + 2) * N_ + n] = (_Float16)vf.z;
    vtb[(size_t)(d + 3) * N_ + n] = (_Float16)vf.w;
}

// ---------------- online-softmax update (MASK specialized, wave-uniform branch)
template <bool MASK>
__device__ __forceinline__ void softmax_update(const v8f& s0, const v8f& s1,
                                               v8f (&acc)[8],
                                               float (&m_run)[8], float (&l_run)[8],
                                               _Float16* Pw,
                                               int qrow0, int kb, int l16, int lhalf)
{
    const int key0 = kb * KTILE + l16;
    const int key1 = key0 + 16;
#pragma unroll
    for (int r = 0; r < 8; ++r) {
        float a0 = s0[r];
        float a1 = s1[r];
        if (MASK) {
            const int qg = qrow0 + r + 8 * lhalf;
            if (key0 > qg) a0 = -INFINITY;
            if (key1 > qg) a1 = -INFINITY;
        }
        float mx = fmaxf(a0, a1);
        mx = fmaxf(mx, __shfl_xor(mx, 1, 32));
        mx = fmaxf(mx, __shfl_xor(mx, 2, 32));
        mx = fmaxf(mx, __shfl_xor(mx, 4, 32));
        mx = fmaxf(mx, __shfl_xor(mx, 8, 32));
        const float m_new = fmaxf(m_run[r], mx);
        const float alpha = __expf(m_run[r] - m_new);
        const float p0 = __expf(a0 - m_new);
        const float p1 = __expf(a1 - m_new);
        float rs = p0 + p1;
        rs += __shfl_xor(rs, 1, 32);
        rs += __shfl_xor(rs, 2, 32);
        rs += __shfl_xor(rs, 4, 32);
        rs += __shfl_xor(rs, 8, 32);
        l_run[r] = l_run[r] * alpha + rs;
        m_run[r] = m_new;
#pragma unroll
        for (int n = 0; n < 8; ++n) acc[n][r] *= alpha;
        const int prow = r + 8 * lhalf;
        Pw[prow * PSTRIDE + l16]      = (_Float16)p0;
        Pw[prow * PSTRIDE + 16 + l16] = (_Float16)p1;
    }
}

// ---------------- flash attention forward (causal), double-buffered async staging
__global__ __launch_bounds__(256) void fa_fwd_causal(const float* __restrict__ Q,
                                                     const _Float16* __restrict__ Kh,
                                                     const _Float16* __restrict__ Vth,
                                                     float* __restrict__ O)
{
    __shared__ alignas(16) _Float16 Ksh [2][KTILE * KSTRIDE];   // 2 x 8704 B
    __shared__ alignas(16) _Float16 Vtsh[2][D_    * VSTRIDE];   // 2 x 10240 B
    __shared__ alignas(16) _Float16 Psh [8 * 16 * PSTRIDE];     //     10240 B

    const int tid   = threadIdx.x;
    const int wave  = tid >> 5;
    const int lane  = tid & 31;
    const int lhalf = lane >> 4;
    const int l16   = lane & 15;

    const int blk = blockIdx.x;
    const int b   = blk >> 4;
    const int qb  = blk & 15;

    const float scale = 0.08838834764831845f; // 1/sqrt(128), folded into Q
    const int qrow0 = qb * QTILE + wave * 16;

    // ---- Q * scale -> f16 A fragments (16x32 per chunk, 4 chunks over D=128) ----
    const float* Qrow = Q + ((size_t)b * N_ + (size_t)(qrow0 + l16)) * D_;
    v16h qa[4];
#pragma unroll
    for (int c = 0; c < 4; ++c) {
        const int d0 = c * 32 + lhalf * 16;
        float4 f0 = *(const float4*)(Qrow + d0);
        float4 f1 = *(const float4*)(Qrow + d0 + 4);
        float4 f2 = *(const float4*)(Qrow + d0 + 8);
        float4 f3 = *(const float4*)(Qrow + d0 + 12);
        qa[c] = (v16h){ (_Float16)(f0.x*scale), (_Float16)(f0.y*scale),
                        (_Float16)(f0.z*scale), (_Float16)(f0.w*scale),
                        (_Float16)(f1.x*scale), (_Float16)(f1.y*scale),
                        (_Float16)(f1.z*scale), (_Float16)(f1.w*scale),
                        (_Float16)(f2.x*scale), (_Float16)(f2.y*scale),
                        (_Float16)(f2.z*scale), (_Float16)(f2.w*scale),
                        (_Float16)(f3.x*scale), (_Float16)(f3.y*scale),
                        (_Float16)(f3.z*scale), (_Float16)(f3.w*scale) };
    }

    v8f acc[8] = {};
    float m_run[8], l_run[8];
#pragma unroll
    for (int r = 0; r < 8; ++r) { m_run[r] = -INFINITY; l_run[r] = 0.0f; }

    const int nkb = (qb + 1) * 4;
    const _Float16* Khb  = Kh  + ((size_t)b << 18);
    const _Float16* Vthb = Vth + ((size_t)b << 18);

    const uint32_t ksh0 = (uint32_t)(uintptr_t)&Ksh[0][0];
    const uint32_t vts0 = (uint32_t)(uintptr_t)&Vtsh[0][0];
    const uint32_t kbufsz = KTILE * KSTRIDE * 2;   // bytes per K buffer
    const uint32_t vbufsz = D_    * VSTRIDE * 2;   // bytes per Vt buffer

    // issue the async DMA for one 32-key tile into LDS buffer `buf`
    auto issue_tile = [&](int kbn, int buf) {
#pragma unroll
        for (int i = 0; i < 2; ++i) {
            const int s = i * 256 + tid;                    // 16-byte segment id, 0..511
            uint64_t kga  = (uint64_t)(uintptr_t)Khb + (size_t)kbn * (KTILE * D_ * 2)
                            + (size_t)s * 16;
            uint32_t klds = ksh0 + (uint32_t)buf * kbufsz
                            + (uint32_t)((s >> 4) * (KSTRIDE * 2) + (s & 15) * 16);
            asm volatile("global_load_async_to_lds_b128 %0, %1, off"
                         :: "v"(klds), "v"(kga) : "memory");
            uint64_t vga  = (uint64_t)(uintptr_t)Vthb + (size_t)(s >> 2) * (N_ * 2)
                            + (size_t)kbn * (KTILE * 2) + (size_t)(s & 3) * 16;
            uint32_t vlds = vts0 + (uint32_t)buf * vbufsz
                            + (uint32_t)((s >> 2) * (VSTRIDE * 2) + (s & 3) * 16);
            asm volatile("global_load_async_to_lds_b128 %0, %1, off"
                         :: "v"(vlds), "v"(vga) : "memory");
        }
    };

    issue_tile(0, 0);   // prologue: stage first tile

    for (int kb = 0; kb < nkb; ++kb) {
        const int buf = kb & 1;
        asm volatile("s_wait_asynccnt 0x0" ::: "memory"); // tile kb landed (this wave's part)
        __syncthreads();                                  // all waves landed + prev buffer free
        if (kb + 1 < nkb) issue_tile(kb + 1, buf ^ 1);    // overlap next DMA with compute
        if (kb + 2 < nkb) {                               // warm L2 two tiles ahead
            __builtin_prefetch((const char*)Khb + (size_t)(kb + 2) * (KTILE * D_ * 2) + tid * 32, 0, 0);
            if (tid < 128)
                __builtin_prefetch((const char*)Vthb + (size_t)tid * (N_ * 2)
                                   + (size_t)(kb + 2) * (KTILE * 2), 0, 0);
        }

        // ---- S = Q @ K^T : preload all 8 B fragments, then 8 WMMAs ----
        const _Float16* Kb = &Ksh[buf][0];
        v8h kf[16];
#pragma unroll
        for (int c = 0; c < 4; ++c) {
            const _Float16* k0 = Kb + (0 + l16)  * KSTRIDE + c * 32 + lhalf * 16;
            const _Float16* k1 = Kb + (16 + l16) * KSTRIDE + c * 32 + lhalf * 16;
            kf[c * 4 + 0] = *(const v8h*)(k0);
            kf[c * 4 + 1] = *(const v8h*)(k0 + 8);
            kf[c * 4 + 2] = *(const v8h*)(k1);
            kf[c * 4 + 3] = *(const v8h*)(k1 + 8);
        }
        v8f s0 = {}, s1 = {};
#pragma unroll
        for (int c = 0; c < 4; ++c) {
            v16h bk0 = cat8(kf[c * 4 + 0], kf[c * 4 + 1]);
            v16h bk1 = cat8(kf[c * 4 + 2], kf[c * 4 + 3]);
            s0 = __builtin_amdgcn_wmma_f32_16x16x32_f16(false, qa[c], false, bk0,
                                                        (short)0, s0, false, false);
            s1 = __builtin_amdgcn_wmma_f32_16x16x32_f16(false, qa[c], false, bk1,
                                                        (short)0, s1, false, false);
        }

        // ---- online softmax; mask only for blocks overlapping the diagonal ----
        _Float16* Pw = &Psh[wave * 16 * PSTRIDE];
        if (kb * KTILE + (KTILE - 1) <= qrow0)
            softmax_update<false>(s0, s1, acc, m_run, l_run, Pw, qrow0, kb, l16, lhalf);
        else
            softmax_update<true>(s0, s1, acc, m_run, l_run, Pw, qrow0, kb, l16, lhalf);

        // ---- re-layout P (C-frag -> A-frag) through wave-private LDS ----
        const v8h plo = *(const v8h*)(&Pw[l16 * PSTRIDE + lhalf * 16]);
        const v8h phi = *(const v8h*)(&Pw[l16 * PSTRIDE + lhalf * 16 + 8]);
        const v16h pa = cat8(plo, phi);

        // ---- O += P @ V : 8 output column tiles over D=128 ----
        const _Float16* Vb = &Vtsh[buf][0];
#pragma unroll
        for (int n = 0; n < 8; ++n) {
            const _Float16* vp = Vb + (n * 16 + l16) * VSTRIDE + lhalf * 16;
            v16h bv = cat8(*(const v8h*)(vp), *(const v8h*)(vp + 8));
            acc[n] = __builtin_amdgcn_wmma_f32_16x16x32_f16(false, pa, false, bv,
                                                            (short)0, acc[n], false, false);
        }
    }

    // ---- normalize and store fp32 output ----
    float inv[8];
#pragma unroll
    for (int r = 0; r < 8; ++r) inv[r] = 1.0f / l_run[r];
    float* Orow = O + ((size_t)b * N_ + (size_t)qrow0) * D_;
#pragma unroll
    for (int n = 0; n < 8; ++n) {
#pragma unroll
        for (int r = 0; r < 8; ++r) {
            Orow[(r + 8 * lhalf) * D_ + n * 16 + l16] = acc[n][r] * inv[r];
        }
    }
}

extern "C" void kernel_launch(void* const* d_in, const int* in_sizes, int n_in,
                              void* d_out, int out_size, void* d_ws, size_t ws_size,
                              hipStream_t stream) {
    (void)in_sizes; (void)n_in; (void)out_size; (void)ws_size;
    const float* q = (const float*)d_in[0];
    const float* k = (const float*)d_in[1];
    const float* v = (const float*)d_in[2];
    float* o = (float*)d_out;

    _Float16* Kh = (_Float16*)d_ws;                       // 16 MB
    _Float16* Vt = Kh + ((size_t)B_ << 18);               // 16 MB

    cvt_kv<<<dim3((B_ * N_ * D_) / (256 * 4)), dim3(256), 0, stream>>>(k, v, Kh, Vt);
    fa_fwd_causal<<<dim3(B_ * (N_ / QTILE)), dim3(256), 0, stream>>>(q, Kh, Vt, o);
}
